// ContConv1dSim_85899345920076
// MI455X (gfx1250) — compile-verified, matrix operand order checked by reading, not run on hardware
//
#include <hip/hip_runtime.h>

// ContConv1dSim fused kernel for gfx1250 (MI455X, wave32, WMMA) — round 3.
//
// Transposed formulation:
//   hT  = W1^T @ te^T        (4x WMMA 16x16x32 f16 per k)
//   kvT = W2^T @ h           (2x WMMA per 16-col tile per k, K=64)
//   out[pos,o] += pcf[pos,i] * kvT[col,pos]
//
// hT's D-layout maps lane-locally onto the next GEMM's B-fragment layout
// (no LDS transpose); in kvT's D-layout each lane owns one position so the
// pcf weight is a register scalar. W1 and W2 are pre-converted to f16 and
// pre-swizzled into per-lane contiguous fragment order in d_ws.
// All masking is branchless (clamped indices + float multiplies) so phase A
// has no EXEC divergence and loads overlap. The b2 contribution is factored
// out of the k loop via prsum = sum_k pcf, leaving one v_fmac per element.

typedef __attribute__((ext_vector_type(16))) _Float16 v16h;
typedef __attribute__((ext_vector_type(8)))  float    v8f;

#define BSZ   8
#define LL    256
#define KK    5
#define CIN   32
#define COUT  32
#define HID   64
#define LEXP  1276            // (SIM+1)*(L-1)+1
#define ROWS  (BSZ * LEXP)    // 10208 = 638 * 16
#define NW    4               // waves per block
#define NCOL  (CIN * COUT)    // 1024 W2 columns
#define W1OFF (64 * 2 * 32 * 16)   // half-offset of W1 fragments in d_ws

// K index (cin or hid-within-chunk) for element e of a 16-bit 16x32 fragment.
__device__ __forceinline__ int fragK(int e, int hs) {
    return ((e >> 3) << 4) | (hs << 3) | (e & 7);
}

// ---- prep: W2 (64x1024) and W1 (32x64) f32 -> f16 fragment-swizzled ws -----
// W2: half index ((tile*2 + c)*32 + lane)*16 + e   (128 KB)
// W1: half index W1OFF + (nt*32 + lane)*16 + e     (  4 KB)
__global__ __launch_bounds__(256)
void prep_weights(const float* __restrict__ W2, const float* __restrict__ W1,
                  _Float16* __restrict__ wsh)
{
    const int t = blockIdx.x * blockDim.x + threadIdx.x;
    if (t < 4096) {                                   // W2 fragments
        const int lane = t & 31;
        const int c    = (t >> 5) & 1;
        const int tile = t >> 6;                      // 0..63
        const int n    = lane & 15;
        const int hs   = lane >> 4;
        v16h frag;
#pragma unroll
        for (int e = 0; e < 16; ++e) {
            int hid = c * 32 + fragK(e, hs);
            frag[e] = (_Float16)W2[hid * NCOL + tile * 16 + n];  // W2^T[col,hid]
        }
        *(v16h*)(wsh + (((size_t)tile * 2 + c) * 32 + lane) * 16) = frag;
    } else if (t < 4096 + 128) {                      // W1^T A-fragments
        const int u    = t - 4096;
        const int lane = u & 31;
        const int nt   = u >> 5;                      // 0..3
        const int n    = lane & 15;
        const int hs   = lane >> 4;
        v16h frag;
#pragma unroll
        for (int e = 0; e < 16; ++e)
            frag[e] = (_Float16)W1[fragK(e, hs) * HID + nt * 16 + n];
        *(v16h*)(wsh + W1OFF + ((size_t)nt * 32 + lane) * 16) = frag;
    }
}

// --------------------------------- main -------------------------------------
__global__ __launch_bounds__(128)
void cc1dsim_fused(const float* __restrict__ times,
                   const float* __restrict__ true_times,
                   const float* __restrict__ true_feat,
                   const unsigned char* __restrict__ npm,   // bool mask (BS,L)
                   const float* __restrict__ b1,
                   const _Float16* __restrict__ wsh,        // prepped W1/W2
                   const float* __restrict__ b2,
                   float* __restrict__ out)
{
    __shared__ float outs[16 * COUT];   // 2 KB cross-wave reduction tile

    const int tid  = threadIdx.x;
    const int w    = tid >> 5;      // wave id: owns cin group [w*8, w*8+8)
    const int lane = tid & 31;
    const int n    = lane & 15;     // position-in-tile owned by this lane
    const int hs   = lane >> 4;

    for (int q = tid; q < 16 * COUT; q += blockDim.x) outs[q] = 0.0f;
    __syncthreads();                // zeroing visible before any ds_add_f32

    const int   r  = blockIdx.x * 16 + n;   // flat row: b*LEXP + j
    const int   b  = r / LEXP;
    const int   j  = r - b * LEXP;
    const int   lq = j / 5;                 // floor window index
    const int   lc = (j + 4) / 5;           // ceil window index (mask)
    const int   mi = b * LL;
    const float tj   = times[r];
    const float npml = (npm[mi + lc] != 0) ? 1.0f : 0.0f;   // k-invariant

    // inverse positional divisors for this lane's fragment elements
    float invp[16];
#pragma unroll
    for (int e = 0; e < 16; ++e) {
        int ci = fragK(e, hs);
        invp[e] = exp2f(-(float)(ci & ~1) * 0.4152410118609203f); // 10000^-(ci&~1)/32
    }

    // W1^T A-fragments (pre-swizzled f16): 4x 32B vector loads
    v16h w1a[4];
#pragma unroll
    for (int nt = 0; nt < 4; ++nt)
        w1a[nt] = *(const v16h*)(wsh + W1OFF + ((size_t)nt * 32 + lane) * 16);

    // b1 for hT D-layout: element (v,lane) is hid = nt*16 + 8*hs + v
    float b1r[4][8];
#pragma unroll
    for (int nt = 0; nt < 4; ++nt) {
        float4 lo = *(const float4*)(b1 + nt * 16 + 8 * hs);
        float4 hi = *(const float4*)(b1 + nt * 16 + 8 * hs + 4);
        b1r[nt][0] = lo.x; b1r[nt][1] = lo.y; b1r[nt][2] = lo.z; b1r[nt][3] = lo.w;
        b1r[nt][4] = hi.x; b1r[nt][5] = hi.y; b1r[nt][6] = hi.z; b1r[nt][7] = hi.w;
    }

    v16h  hB[KK][2];   // h as B-fragments (K=64 in two chunks), all k live
    float pr[KK][8];   // pcf[pos=n, i = w*8 + t2] weights, all k live

    // ------- Phase A: te -> hT (branchless, register-only reshuffle) -------
    for (int k = 0; k < KK; ++k) {
        const int src = lq + k;
        const int x   = lc + k;
        const int xi  = (x - KK)   > 0 ? (x - KK)   : 0;   // clamped, in-bounds
        const int si  = (src - KK) > 0 ? (src - KK) : 0;

        // branchless masks as float multipliers (no EXEC divergence)
        float mpf  = (x >= KK)   ? 1.0f : 0.0f;
        mpf       *= (npm[mi + xi] != 0) ? 1.0f : 0.0f;
        const float dtmf = mpf * npml;                       // dt_mask
        const float svf  = (src >= KK) ? 1.0f : 0.0f;        // src valid
        const float pct  = true_times[mi + si] * svf;        // unconditional load
        const float delta = (tj - pct) * dtmf;
        const float fvf   = dtmf * svf;                      // pcf mask

        // te^T B-fragment: lane col = position n, element -> cin
        v16h a;
#pragma unroll
        for (int e = 0; e < 16; ++e) {
            float ang = delta * invp[e];
            a[e] = (_Float16)((e & 1) ? __cosf(ang) : __sinf(ang));
        }

        // hT = relu(W1^T @ te^T + b1): 4 hid-tiles of 16
        v8f accT[4];
#pragma unroll
        for (int nt = 0; nt < 4; ++nt) {
            v8f z = {};
            accT[nt] = __builtin_amdgcn_wmma_f32_16x16x32_f16(
                false, w1a[nt], false, a, (short)0, z, false, false);
        }

        // D-layout -> B-fragment layout is lane-local:
        //   hid = c*32 + fragK(e,hs)  ==  nt = c*2 + (e>>3), v = e&7
#pragma unroll
        for (int c = 0; c < 2; ++c)
#pragma unroll
            for (int e = 0; e < 16; ++e) {
                int nt = c * 2 + (e >> 3);
                int v  = e & 7;
                float hv = fmaxf(accT[nt][v] + b1r[nt][v], 0.0f);
                hB[k][c][e] = (_Float16)hv;
            }

        // pcf weights: unconditional vector load, masked by multiply
        {
            const float4* fp =
                (const float4*)(true_feat + (size_t)(mi + si) * CIN + w * 8);
            float4 f0 = fp[0], f1 = fp[1];
            pr[k][0] = f0.x * fvf; pr[k][1] = f0.y * fvf;
            pr[k][2] = f0.z * fvf; pr[k][3] = f0.w * fvf;
            pr[k][4] = f1.x * fvf; pr[k][5] = f1.y * fvf;
            pr[k][6] = f1.z * fvf; pr[k][7] = f1.w * fvf;
        }
    }

    // prsum[t2] = sum_k pcf  (factors the b2 term out of the k loop)
    float prsum[8];
#pragma unroll
    for (int t2 = 0; t2 < 8; ++t2) {
        float s = pr[0][t2];
#pragma unroll
        for (int k = 1; k < KK; ++k) s += pr[k][t2];
        prsum[t2] = s;
    }

    // ---- Phase B: kvT = W2^T @ h, fold pcf immediately (wave: 16 tiles) ----
    float outAcc[2][8] = {};

    for (int t2 = 0; t2 < 8; ++t2) {
#pragma unroll
        for (int p = 0; p < 2; ++p) {
            const int tile = w * 16 + t2 * 2 + p;     // 16-col W2 tile
            const int cb   = tile * 16;

            // pre-swizzled f16 fragments: 2x 32B contiguous per lane
            const v16h f0 = *(const v16h*)(wsh + (((size_t)tile * 2 + 0) * 32 + lane) * 16);
            const v16h f1 = *(const v16h*)(wsh + (((size_t)tile * 2 + 1) * 32 + lane) * 16);

            for (int k = 0; k < KK; ++k) {
                v8f acc = {};
                acc = __builtin_amdgcn_wmma_f32_16x16x32_f16(
                    false, f0, false, hB[k][0], (short)0, acc, false, false);
                acc = __builtin_amdgcn_wmma_f32_16x16x32_f16(
                    false, f1, false, hB[k][1], (short)0, acc, false, false);
                const float pw = pr[k][t2];
#pragma unroll
                for (int v = 0; v < 8; ++v)
                    outAcc[p][v] += pw * acc[v];      // single v_fmac each
            }

            // b2 term: (sum_k pcf) * b2[col],  col = cb + 8*hs + v
            float4 b2lo = *(const float4*)(b2 + cb + 8 * hs);
            float4 b2hi = *(const float4*)(b2 + cb + 8 * hs + 4);
            const float ps = prsum[t2];
            outAcc[p][0] += ps * b2lo.x; outAcc[p][1] += ps * b2lo.y;
            outAcc[p][2] += ps * b2lo.z; outAcc[p][3] += ps * b2lo.w;
            outAcc[p][4] += ps * b2hi.x; outAcc[p][5] += ps * b2hi.y;
            outAcc[p][6] += ps * b2hi.z; outAcc[p][7] += ps * b2hi.w;
        }
    }

    // --------------- cross-wave (cin-split) reduction + store ---------------
#pragma unroll
    for (int p = 0; p < 2; ++p)
#pragma unroll
        for (int v = 0; v < 8; ++v)
            atomicAdd(&outs[n * COUT + p * 16 + 8 * hs + v], outAcc[p][v]);
    __syncthreads();

    const int base = tid * 4;      // 512 contiguous f32 per block
    *(float4*)(out + (size_t)blockIdx.x * 512 + base) = *(const float4*)&outs[base];
}

extern "C" void kernel_launch(void* const* d_in, const int* in_sizes, int n_in,
                              void* d_out, int out_size, void* d_ws, size_t ws_size,
                              hipStream_t stream) {
    (void)in_sizes; (void)n_in; (void)out_size; (void)ws_size;
    const float*         times = (const float*)d_in[0];
    const float*         tt    = (const float*)d_in[1];
    const float*         tf    = (const float*)d_in[2];
    const unsigned char* npm   = (const unsigned char*)d_in[3];
    const float*         W1    = (const float*)d_in[4];
    const float*         b1    = (const float*)d_in[5];
    const float*         W2    = (const float*)d_in[6];
    const float*         b2    = (const float*)d_in[7];
    float*               out   = (float*)d_out;
    _Float16*            wsh   = (_Float16*)d_ws;   // needs 132 KB scratch

    prep_weights<<<17, 256, 0, stream>>>(W2, W1, wsh);        // W2 + W1 swizzle
    cc1dsim_fused<<<ROWS / 16, 32 * NW, 0, stream>>>(times, tt, tf, npm,
                                                     b1, wsh, b2, out);
}
